// TUPEMultiheadAttention_13743895347549
// MI455X (gfx1250) — matrix operand order repaired and checked by reading, not compiled, not run on hardware
//
#include <hip/hip_runtime.h>
#include <hip/hip_bf16.h>

#define SDIM 2048
#define DDIM 1024
#define HHH  16
#define BBB  4
// head dim 64, concat qk dim 128

typedef __bf16 v16bf __attribute__((ext_vector_type(16)));
typedef float  v8f   __attribute__((ext_vector_type(8)));

union Frag {
  v16bf v;
  uint4 q[2];
  unsigned short u[16];
};

__device__ __forceinline__ unsigned short f2bf(float f) {
  unsigned int x = __float_as_uint(f);
  x += 0x7FFFu + ((x >> 16) & 1u);
  return (unsigned short)(x >> 16);
}

__device__ __forceinline__ v8f wmma_bf16(const Frag& a, const Frag& b, v8f c) {
  return __builtin_amdgcn_wmma_f32_16x16x32_bf16(false, a.v, false, b.v,
                                                 (short)0, c, false, false);
}

// ---------------- fp32 -> bf16 conversion (vectorized x4) ----------------
__global__ void cvt_f32_bf16_x4(const float* __restrict__ in,
                                unsigned short* __restrict__ out, int n4) {
  int i = blockIdx.x * blockDim.x + threadIdx.x;
  if (i >= n4) return;
  float4 f = reinterpret_cast<const float4*>(in)[i];
  uint2 o;
  o.x = (unsigned)f2bf(f.x) | ((unsigned)f2bf(f.y) << 16);
  o.y = (unsigned)f2bf(f.z) | ((unsigned)f2bf(f.w) << 16);
  reinterpret_cast<uint2*>(out)[i] = o;
}

// ---------------- WMMA GEMM: C[M,N] = A[M,1024] * W[N,1024]^T ----------------
// MODE 0: A=x,  W=Wqkv (N=3072) -> scatter q->Qcat[0:64], k->Kcat[0:64], v->Vt (transposed)
// MODE 1: A=PE, W=UqUk (N=2048) -> scatter Uq->Qcat[64:128], Uk->Kcat[64:128]
// MODE 2: A=values, W=Wo (N=1024) -> fp32 Out
template <int MODE>
__global__ __launch_bounds__(128) void gemm_bf16_wmma(
    const unsigned short* __restrict__ A,
    const unsigned short* __restrict__ W,
    unsigned short* __restrict__ Qcat,
    unsigned short* __restrict__ Kcat,
    unsigned short* __restrict__ Vt,
    float* __restrict__ Out) {
  const int KK   = DDIM;
  const int lane = threadIdx.x & 31;
  const int wave = threadIdx.x >> 5;
  const int col  = lane & 15;
  const int half = lane >> 4;
  const int m0 = blockIdx.x * 16;
  const int n0 = (blockIdx.y * 4 + wave) * 64;

  const unsigned short* arow = A + (size_t)(m0 + col) * KK;

  v8f C[4];
#pragma unroll
  for (int i = 0; i < 4; ++i)
#pragma unroll
    for (int r = 0; r < 8; ++r) C[i][r] = 0.0f;

  for (int k = 0; k < KK; k += 32) {
    // A fragment: 16x32 bf16, lane row = col, K = 16*half .. 16*half+15 contiguous
    Frag Af;
    Af.q[0] = *reinterpret_cast<const uint4*>(arow + k + 16 * half);
    Af.q[1] = *reinterpret_cast<const uint4*>(arow + k + 16 * half + 8);
#pragma unroll
    for (int c4 = 0; c4 < 4; ++c4) {
      // B fragment: 32x16 bf16, lane column = col, K = 16*half .. 16*half+15 contiguous
      const unsigned short* wrow = W + (size_t)(n0 + c4 * 16 + col) * KK + k + 16 * half;
      Frag Bf;
      Bf.q[0] = *reinterpret_cast<const uint4*>(wrow);
      Bf.q[1] = *reinterpret_cast<const uint4*>(wrow + 8);
      C[c4] = wmma_bf16(Af, Bf, C[c4]);
    }
  }

#pragma unroll
  for (int c4 = 0; c4 < 4; ++c4) {
#pragma unroll
    for (int r = 0; r < 8; ++r) {
      const int m = m0 + r + 8 * half;   // C/D layout: M = r + 8*(lane>>4)
      const int n = n0 + c4 * 16 + col;  // N = lane & 15
      const float val = C[c4][r];
      if (MODE == 2) {
        Out[(size_t)m * DDIM + n] = val;
      } else {
        const int b = m >> 11;
        const int s = m & 2047;
        const unsigned short bv = f2bf(val);
        if (MODE == 0) {
          const int h = n / 192;
          const int c = n % 192;
          const size_t bh = (size_t)(b * HHH + h);
          if (c < 64)       Qcat[(bh * SDIM + s) * 128 + c] = bv;
          else if (c < 128) Kcat[(bh * SDIM + s) * 128 + (c - 64)] = bv;
          else              Vt[(bh * 64 + (c - 128)) * SDIM + s] = bv;  // V transposed
        } else {
          const int h = n >> 7;
          const int c = n & 127;
          const size_t bh = (size_t)(b * HHH + h);
          if (c < 64) Qcat[(bh * SDIM + s) * 128 + 64 + c] = bv;   // Uq
          else        Kcat[(bh * SDIM + s) * 128 + c] = bv;        // Uk
        }
      }
    }
  }
}

// ---------------- Fused flash attention (transposed-score trick) ----------------
// One wave per 16-query tile; block = 4 waves = 64 queries. Per 32-key block:
// S^T = K_tile(A) x Q^T(B): 8 wmma; softmax in-lane; P repack in-lane; P x V: 4 wmma.
__global__ __launch_bounds__(128) void attn_fused_wmma(
    const unsigned short* __restrict__ Qcat,
    const unsigned short* __restrict__ Kcat,
    const unsigned short* __restrict__ Vt,
    const float* __restrict__ PEr,
    unsigned short* __restrict__ Vals) {
  const int lane = threadIdx.x & 31;
  const int wave = threadIdx.x >> 5;
  const int col  = lane & 15;
  const int half = lane >> 4;
  const int bh = blockIdx.x;
  const int b  = bh >> 4;
  const int h  = bh & 15;
  const int q0 = blockIdx.y * 64 + wave * 16;

  const unsigned short* Qb = Qcat + (size_t)bh * SDIM * 128;
  const unsigned short* Kb = Kcat + (size_t)bh * SDIM * 128;
  const unsigned short* Vb = Vt   + (size_t)bh * 64 * SDIM;
  const float* Bias = PEr + (size_t)h * SDIM * SDIM + (size_t)(q0 + col) * SDIM;

  // Q^T fragments (B operand): lane = query column, K(features) = 16*half..+15
  Frag Qf[4];
  {
    const unsigned short* qrow = Qb + (size_t)(q0 + col) * 128;
#pragma unroll
    for (int j = 0; j < 4; ++j) {
      Qf[j].q[0] = *reinterpret_cast<const uint4*>(qrow + j * 32 + 16 * half);
      Qf[j].q[1] = *reinterpret_cast<const uint4*>(qrow + j * 32 + 16 * half + 8);
    }
  }

  v8f O[4];
#pragma unroll
  for (int i = 0; i < 4; ++i)
#pragma unroll
    for (int r = 0; r < 8; ++r) O[i][r] = 0.0f;

  float m = -3.0e38f;
  float ssum = 0.0f;
  const float scale = 0.08838834764831845f;  // 1/sqrt(128)

  for (int kb = 0; kb < SDIM; kb += 32) {
    float sv[2][8];
    float rmax = -3.0e38f;
#pragma unroll
    for (int t = 0; t < 2; ++t) {
      // K fragments (A operand): lane row = key (col), K = 16*half..+15
      const unsigned short* krow = Kb + (size_t)(kb + t * 16 + col) * 128;
      v8f c;
#pragma unroll
      for (int r = 0; r < 8; ++r) c[r] = 0.0f;
#pragma unroll
      for (int j = 0; j < 4; ++j) {
        Frag Kf;
        Kf.q[0] = *reinterpret_cast<const uint4*>(krow + j * 32 + 16 * half);
        Kf.q[1] = *reinterpret_cast<const uint4*>(krow + j * 32 + 16 * half + 8);
        c = wmma_bf16(Kf, Qf[j], c);  // S^T tile: M=key, N=query
      }
      // bias: S^T C-layout lane holds keys (kb+16t+8h+r), query q0+col -> 8 consecutive floats
      const float* bp = Bias + kb + t * 16 + 8 * half;
      __builtin_prefetch(bp + 32, 0, 1);
      float4 b0 = *reinterpret_cast<const float4*>(bp);
      float4 b1 = *reinterpret_cast<const float4*>(bp + 4);
      const float bb[8] = {b0.x, b0.y, b0.z, b0.w, b1.x, b1.y, b1.z, b1.w};
#pragma unroll
      for (int r = 0; r < 8; ++r) {
        const float s = c[r] * scale + bb[r];
        sv[t][r] = s;
        rmax = fmaxf(rmax, s);
      }
    }
    // row (query) max across the two half-wave key groups
    rmax = fmaxf(rmax, __shfl_xor(rmax, 16, 32));
    const float mnew = fmaxf(m, rmax);
    const float corr = __expf(m - mnew);
    float bsum = 0.0f;
#pragma unroll
    for (int t = 0; t < 2; ++t)
#pragma unroll
      for (int r = 0; r < 8; ++r) {
        const float p = __expf(sv[t][r] - mnew);
        sv[t][r] = p;
        bsum += p;
      }
    bsum += __shfl_xor(bsum, 16, 32);
    ssum = ssum * corr + bsum;
    m = mnew;

    // rescale O: remap per-query corr (keyed by lane&15) to O rows (r + 8*half)
    float corrO[8];
#pragma unroll
    for (int r = 0; r < 8; ++r) corrO[r] = __shfl(corr, r + 8 * half, 32);
#pragma unroll
    for (int c4 = 0; c4 < 4; ++c4)
#pragma unroll
      for (int r = 0; r < 8; ++r) O[c4][r] *= corrO[r];

    // repack P into A-fragment (16 queries x 32 keys) -- purely in-lane
    Frag Pf;
#pragma unroll
    for (int j = 0; j < 8; ++j) {
      Pf.u[j]     = f2bf(sv[0][j]);  // keys 8*half + 0..7
      Pf.u[8 + j] = f2bf(sv[1][j]);  // keys 16 + 8*half + 0..7
    }

    // P x V: B operand from transposed V, contiguous along keys
#pragma unroll
    for (int c4 = 0; c4 < 4; ++c4) {
      const unsigned short* vrow = Vb + (size_t)(c4 * 16 + col) * SDIM + kb + 8 * half;
      Frag Vf;
      Vf.q[0] = *reinterpret_cast<const uint4*>(vrow);
      Vf.q[1] = *reinterpret_cast<const uint4*>(vrow + 16);
      O[c4] = wmma_bf16(Pf, Vf, O[c4]);
    }
  }

  const float inv = 1.0f / ssum;
  float invO[8];
#pragma unroll
  for (int r = 0; r < 8; ++r) invO[r] = __shfl(inv, r + 8 * half, 32);
#pragma unroll
  for (int c4 = 0; c4 < 4; ++c4) {
#pragma unroll
    for (int r = 0; r < 8; ++r) {
      const int q  = q0 + r + 8 * half;
      const int cc = c4 * 16 + col;
      Vals[((size_t)b * SDIM + q) * DDIM + h * 64 + cc] = f2bf(O[c4][r] * invO[r]);
    }
  }
}

// ---------------- host launch ----------------
extern "C" void kernel_launch(void* const* d_in, const int* in_sizes, int n_in,
                              void* d_out, int out_size, void* d_ws, size_t ws_size,
                              hipStream_t stream) {
  (void)in_sizes; (void)n_in; (void)out_size; (void)ws_size;
  const float* x    = (const float*)d_in[0];
  const float* PE   = (const float*)d_in[1];
  const float* PEr  = (const float*)d_in[2];
  const float* Wqkv = (const float*)d_in[3];
  const float* UqUk = (const float*)d_in[4];
  const float* Wo   = (const float*)d_in[5];
  float* out = (float*)d_out;

  const size_t nX    = (size_t)BBB * SDIM * DDIM;   // 8388608
  const size_t nWqkv = (size_t)3 * DDIM * DDIM;     // 3145728
  const size_t nUqUk = (size_t)2 * DDIM * DDIM;     // 2097152
  const size_t nWo   = (size_t)DDIM * DDIM;         // 1048576
  const size_t nQK   = (size_t)BBB * HHH * SDIM * 128;  // 16777216
  const size_t nVt   = (size_t)BBB * HHH * 64 * SDIM;   // 8388608

  char* ws = (char*)d_ws;
  size_t off = 0;
  auto take = [&](size_t elems) {
    unsigned short* p = (unsigned short*)(ws + off);
    off = (off + elems * 2 + 255) & ~(size_t)255;
    return p;
  };
  unsigned short* xb    = take(nX);
  unsigned short* PEb   = take(nX);
  unsigned short* Wqkvb = take(nWqkv);
  unsigned short* UqUkb = take(nUqUk);
  unsigned short* Wob   = take(nWo);
  unsigned short* Qcat  = take(nQK);
  unsigned short* Kcat  = take(nQK);
  unsigned short* Vt    = take(nVt);
  unsigned short* Vals  = take(nX);

  const int thr = 256;
  cvt_f32_bf16_x4<<<(int)(nX / 4 / thr),    thr, 0, stream>>>(x, xb, (int)(nX / 4));
  cvt_f32_bf16_x4<<<(int)(nX / 4 / thr),    thr, 0, stream>>>(PE, PEb, (int)(nX / 4));
  cvt_f32_bf16_x4<<<(int)(nWqkv / 4 / thr), thr, 0, stream>>>(Wqkv, Wqkvb, (int)(nWqkv / 4));
  cvt_f32_bf16_x4<<<(int)(nUqUk / 4 / thr), thr, 0, stream>>>(UqUk, UqUkb, (int)(nUqUk / 4));
  cvt_f32_bf16_x4<<<(int)(nWo / 4 / thr),   thr, 0, stream>>>(Wo, Wob, (int)(nWo / 4));

  // M tiles = 8192/16 = 512; N per block = 256 (4 waves x 64)
  gemm_bf16_wmma<0><<<dim3(512, 12), 128, 0, stream>>>(xb,  Wqkvb, Qcat, Kcat, Vt, nullptr);
  gemm_bf16_wmma<1><<<dim3(512, 8),  128, 0, stream>>>(PEb, UqUkb, Qcat, Kcat, Vt, nullptr);

  attn_fused_wmma<<<dim3(BBB * HHH, SDIM / 64), 128, 0, stream>>>(Qcat, Kcat, Vt, PEr, Vals);

  gemm_bf16_wmma<2><<<dim3(512, 4), 128, 0, stream>>>(Vals, Wob, nullptr, nullptr, nullptr, out);
}
